// Fastformer_35124242547158
// MI455X (gfx1250) — matrix operand order repaired and checked by reading, not compiled, not run on hardware
//
#include <hip/hip_runtime.h>
#include <hip/hip_bf16.h>

typedef __attribute__((ext_vector_type(16))) __bf16 v16bf;
typedef __attribute__((ext_vector_type(8)))  float  v8f;

#define Bb 8
#define Ss 2048
#define Hh 16
#define DHd 64
#define Dd 1024

// ===========================================================================
// Elementwise f32 -> bf16 convert (8 elems / thread)
// ===========================================================================
__global__ __launch_bounds__(256) void conv_bf16(
    const float* __restrict__ in, __bf16* __restrict__ out) {
  const long i0 = ((long)blockIdx.x * 256 + threadIdx.x) * 8;
  union { float4 v[2]; float f[8]; } a;
  a.v[0] = *(const float4*)(in + i0);
  a.v[1] = *(const float4*)(in + i0 + 4);
  union { __bf16 h[8]; uint4 u; } o;
#pragma unroll
  for (int j = 0; j < 8; ++j) o.h[j] = (__bf16)a.f[j];
  *(uint4*)(out + i0) = o.u;
}

// X[b,s,d] = bf16( Qf[b,s,d] * gk[b,d] )   (8 elems / thread)
__global__ __launch_bounds__(256) void scale_conv_bf16(
    const float* __restrict__ Qf, const float* __restrict__ gk,
    __bf16* __restrict__ X) {
  const long i0 = ((long)blockIdx.x * 256 + threadIdx.x) * 8;
  const int d = (int)(i0 & (Dd - 1));
  const int b = (int)(i0 >> 21);  // (i0/1024)/2048
  union { float4 v[2]; float f[8]; } a, s;
  a.v[0] = *(const float4*)(Qf + i0);
  a.v[1] = *(const float4*)(Qf + i0 + 4);
  s.v[0] = *(const float4*)(gk + b * Dd + d);
  s.v[1] = *(const float4*)(gk + b * Dd + d + 4);
  union { __bf16 h[8]; uint4 u; } o;
#pragma unroll
  for (int j = 0; j < 8; ++j) o.h[j] = (__bf16)(a.f[j] * s.f[j]);
  *(uint4*)(X + i0) = o.u;
}

// Wt[n*K + k] = bf16( W[k*N + n] )  -- tiled transpose, W is K x N
__global__ __launch_bounds__(256) void transpose_conv_bf16(
    const float* __restrict__ W, __bf16* __restrict__ Wt, int N, int K) {
  __shared__ float t[32][33];
  const int tx = threadIdx.x, ty = threadIdx.y;
#pragma unroll
  for (int j = 0; j < 4; ++j) {
    const int n = blockIdx.x * 32 + tx;
    const int k = blockIdx.y * 32 + ty + j * 8;
    t[ty + j * 8][tx] = W[(long)k * N + n];
  }
  __syncthreads();
#pragma unroll
  for (int j = 0; j < 4; ++j) {
    const int k = blockIdx.y * 32 + tx;
    const int n = blockIdx.x * 32 + ty + j * 8;
    Wt[(long)n * K + k] = (__bf16)t[tx][ty + j * 8];
  }
}

// ===========================================================================
// Big GEMM: C[M,N](f32) = A(M,K bf16) @ Wt(N,K bf16)^T, optional +Cadd.
// BM=128, BN=256, BK=64. 8 waves (2M x 4N), wave tile 64x64 -> 32 WMMA/K-tile
// with only 16 fragment loads (2 WMMA per ds_load pair).
// Double-buffered LDS filled with global_load_async_to_lds_b128 (ASYNCcnt).
// ===========================================================================
#define LDK 72  // halves per LDS row (64 data + 8 pad); 144B, 16B-aligned
#define ABH (128 * LDK)         // A region, halves
#define BBH (256 * LDK)         // B region, halves
#define BUFH (ABH + BBH)        // per-buffer halves

template <bool ADD_C>
__global__ __launch_bounds__(256) void gemm_bf16_async(
    const __bf16* __restrict__ A, const __bf16* __restrict__ Wt,
    const float* __restrict__ cadd, float* __restrict__ C,
    int M, int N, int K) {
  __shared__ __bf16 sm[2 * BUFH];  // [buf][ A:128*LDK | B:256*LDK ]

  const int tid = threadIdx.x;
  const int m0 = blockIdx.y * 128;
  const int n0 = blockIdx.x * 256;

  const int w    = tid >> 5;
  const int lane = tid & 31;
  const int wm   = (w & 1) * 64;
  const int wn   = (w >> 1) * 64;
  const int la   = lane & 15;
  const int kh   = lane >> 4;

  const __bf16* Atile = A + (long)m0 * K;   // 128 rows, K-contiguous
  const __bf16* Btile = Wt + (long)n0 * K;  // 256 rows, K-contiguous
  const uint smBase = (uint)(uintptr_t)&sm[0];

  // async-copy a (rows x 64) bf16 tile into LDS region (16B chunks per lane)
  auto stageA = [&](uint regionB, int kk) {
#pragma unroll
    for (int i = 0; i < 4; ++i) {          // 128 rows * 8 chunks = 1024
      const int c   = tid + i * 256;
      const int row = c >> 3;
      const int col = c & 7;
      const __bf16* g = Atile + (long)row * K + kk + col * 8;
      const uint lds  = regionB + row * (LDK * 2) + col * 16;
      asm volatile("global_load_async_to_lds_b128 %0, %1, off"
                   :: "v"(lds), "v"(g) : "memory");
    }
  };
  auto stageB = [&](uint regionB, int kk) {
#pragma unroll
    for (int i = 0; i < 8; ++i) {          // 256 rows * 8 chunks = 2048
      const int c   = tid + i * 256;
      const int row = c >> 3;
      const int col = c & 7;
      const __bf16* g = Btile + (long)row * K + kk + col * 8;
      const uint lds  = regionB + row * (LDK * 2) + col * 16;
      asm volatile("global_load_async_to_lds_b128 %0, %1, off"
                   :: "v"(lds), "v"(g) : "memory");
    }
  };

  v8f acc[4][4];
#pragma unroll
  for (int i = 0; i < 4; ++i)
#pragma unroll
    for (int j = 0; j < 4; ++j)
      acc[i][j] = (v8f){0.f, 0.f, 0.f, 0.f, 0.f, 0.f, 0.f, 0.f};

  // prologue: fill buffer 0
  stageA(smBase, 0);
  stageB(smBase + ABH * 2, 0);
  asm volatile("s_wait_asynccnt 0x0" ::: "memory");
  __syncthreads();

  int buf = 0;
  for (int kk = 0; kk < K; kk += 64) {
    const int nxt = kk + 64;
    if (nxt < K) {  // prefetch next K-tile into the other buffer
      const uint nb = smBase + (buf ^ 1) * (BUFH * 2);
      stageA(nb, nxt);
      stageB(nb + ABH * 2, nxt);
    }

    const __bf16* bufA = sm + buf * BUFH;
    const __bf16* bufB = bufA + ABH;
#pragma unroll
    for (int ks = 0; ks < 64; ks += 32) {
      v16bf fA[4];
#pragma unroll
      for (int i = 0; i < 4; ++i) {
        const __bf16* p = bufA + (wm + i * 16 + la) * LDK + ks + kh * 8;
        union { uint4 u[2]; v16bf v; } cv;
        cv.u[0] = *(const uint4*)p;          // K 0-7  / 8-15  (this kstep)
        cv.u[1] = *(const uint4*)(p + 16);   // K 16-23 / 24-31
        fA[i] = cv.v;
      }
#pragma unroll
      for (int j = 0; j < 4; ++j) {
        const __bf16* p = bufB + (wn + j * 16 + la) * LDK + ks + kh * 8;
        union { uint4 u[2]; v16bf v; } cv;
        cv.u[0] = *(const uint4*)p;          // K 0-7  / 8-15  (this kstep)
        cv.u[1] = *(const uint4*)(p + 16);   // K 16-23 / 24-31
        const v16bf fB = cv.v;
#pragma unroll
        for (int i = 0; i < 4; ++i)
          acc[i][j] = __builtin_amdgcn_wmma_f32_16x16x32_bf16(
              false, fA[i], false, fB, (short)0, acc[i][j], false, false);
      }
    }

    if (nxt < K) asm volatile("s_wait_asynccnt 0x0" ::: "memory");
    __syncthreads();
    buf ^= 1;
  }

  // epilogue: lane l holds N=l%16, rows M = v + 8*(l/16)
#pragma unroll
  for (int i = 0; i < 4; ++i)
#pragma unroll
    for (int j = 0; j < 4; ++j) {
      const int n = n0 + wn + j * 16 + la;
#pragma unroll
      for (int v = 0; v < 8; ++v) {
        const long m = m0 + wm + i * 16 + kh * 8 + v;
        float val = acc[i][j][v];
        if (ADD_C) val += cadd[m * N + n];
        C[m * N + n] = val;
      }
    }
}

// ===========================================================================
// Thin GEMM: Lg[(b*16+h)*S+s] = 0.125 * (X[row,:].W[:,h]) - (1-mask)*1e8
// One 16x16 WMMA accumulator per wave (N = 16 heads), 8 waves / block.
// ===========================================================================
__global__ __launch_bounds__(256) void logits_wmma(
    const float* __restrict__ X, const float* __restrict__ W,
    const float* __restrict__ mask, float* __restrict__ Lg) {
  __shared__ __bf16 lW[16 * 1040];  // W transposed: [h][k]

  const int tid = threadIdx.x;
  {
    const int h = tid & 15;
    const int kb = (tid >> 4) * 64;
#pragma unroll 4
    for (int j = 0; j < 64; ++j) lW[h * 1040 + kb + j] = (__bf16)W[(long)(kb + j) * Hh + h];
  }
  __syncthreads();

  const int w    = tid >> 5;
  const int lane = tid & 31;
  const int la   = lane & 15;
  const int kh   = lane >> 4;
  const int m0   = blockIdx.x * 128 + w * 16;

  v8f acc = (v8f){0.f, 0.f, 0.f, 0.f, 0.f, 0.f, 0.f, 0.f};
  for (int kk = 0; kk < Dd; kk += 32) {
    const float* p = X + (long)(m0 + la) * Dd + kk + kh * 8;
    union { float4 v[4]; float f[16]; } fa;
    fa.v[0] = *(const float4*)(p + 0);
    fa.v[1] = *(const float4*)(p + 4);
    fa.v[2] = *(const float4*)(p + 16);
    fa.v[3] = *(const float4*)(p + 20);
    union { __bf16 h[16]; v16bf v; } pa;
#pragma unroll
    for (int j = 0; j < 16; ++j) pa.h[j] = (__bf16)fa.f[j];

    const int base = la * 1040 + kk + kh * 8;
    union { uint4 u[2]; v16bf v; } pb;
    pb.u[0] = *(const uint4*)&lW[base];
    pb.u[1] = *(const uint4*)&lW[base + 16];

    acc = __builtin_amdgcn_wmma_f32_16x16x32_bf16(false, pa.v, false, pb.v,
                                                  (short)0, acc, false, false);
  }
#pragma unroll
  for (int v = 0; v < 8; ++v) {
    const int srow = m0 + kh * 8 + v;
    const int b = srow >> 11;
    const int s = srow & (Ss - 1);
    const float val = acc[v] * 0.125f - (1.0f - mask[b * Ss + s]) * 1e8f;
    Lg[((long)(b * Hh + la)) * Ss + s] = val;
  }
}

// ===========================================================================
// Row softmax over S=2048 (one block per (b,h) row), in place.
// ===========================================================================
__global__ __launch_bounds__(256) void softmax_rows(float* __restrict__ Lg) {
  __shared__ float red[256];
  float* row = Lg + (long)blockIdx.x * Ss;
  const int tid = threadIdx.x;
  float v[8];
  float mx = -3.4e38f;
#pragma unroll
  for (int i = 0; i < 8; ++i) { v[i] = row[tid + i * 256]; mx = fmaxf(mx, v[i]); }
  red[tid] = mx;
  __syncthreads();
  for (int s = 128; s > 0; s >>= 1) {
    if (tid < s) red[tid] = fmaxf(red[tid], red[tid + s]);
    __syncthreads();
  }
  const float m = red[0];
  __syncthreads();
  float sum = 0.f;
#pragma unroll
  for (int i = 0; i < 8; ++i) { v[i] = __expf(v[i] - m); sum += v[i]; }
  red[tid] = sum;
  __syncthreads();
  for (int s = 128; s > 0; s >>= 1) {
    if (tid < s) red[tid] += red[tid + s];
    __syncthreads();
  }
  const float inv = 1.0f / red[0];
#pragma unroll
  for (int i = 0; i < 8; ++i) row[tid + i * 256] = v[i] * inv;
}

// ===========================================================================
// g[b*1024+h*64+dh] = sum_s att[(b*16+h)*S+s] * X[b*sb + h*sh + s*ss + dh]
// ===========================================================================
__global__ __launch_bounds__(256) void weighted_sum(
    const float* __restrict__ att, const float* __restrict__ X,
    long sb, long sh, long ss, float* __restrict__ g) {
  const int bh = blockIdx.x;
  const int b = bh >> 4, h = bh & 15;
  const int tid = threadIdx.x;
  const int dh = tid & 63, st = tid >> 6;
  const float* arow = att + (long)bh * Ss;
  const float* xp = X + (long)b * sb + (long)h * sh + dh;
  float acc = 0.f;
  for (int s = st; s < Ss; s += 4) acc += arow[s] * xp[(long)s * ss];
  __shared__ float red[256];
  red[tid] = acc;
  __syncthreads();
  if (st == 0) g[b * Dd + h * DHd + dh] = red[dh] + red[64 + dh] + red[128 + dh] + red[192 + dh];
}

// ===========================================================================
// QK[(b,h,s,dh)] = Kf[b,s,h*64+dh] * gq[b, h*64+dh]
// ===========================================================================
__global__ __launch_bounds__(256) void qk_scale(
    const float* __restrict__ Kf, const float* __restrict__ gq, float* __restrict__ QK) {
  const int vid = blockIdx.x * 256 + threadIdx.x;
  const int dh4 = vid & 15;
  int r = vid >> 4;
  const int s = r & (Ss - 1); r >>= 11;
  const int h = r & 15;
  const int b = r >> 4;
  const int col = h * DHd + dh4 * 4;
  const float4 k = *(const float4*)(Kf + ((long)(b * Ss + s)) * Dd + col);
  const float4 q = *(const float4*)(gq + b * Dd + col);
  float4 o; o.x = k.x * q.x; o.y = k.y * q.y; o.z = k.z * q.z; o.w = k.w * q.w;
  *(float4*)(QK + (long)vid * 4) = o;
}

// ===========================================================================
extern "C" void kernel_launch(void* const* d_in, const int* in_sizes, int n_in,
                              void* d_out, int out_size, void* d_ws, size_t ws_size,
                              hipStream_t stream) {
  const float* Q_seq  = (const float*)d_in[0];
  const float* K_seq  = (const float*)d_in[1];
  const float* Q_mask = (const float*)d_in[2];
  const float* K_mask = (const float*)d_in[3];
  const float* WQ     = (const float*)d_in[4];
  const float* WK     = (const float*)d_in[5];
  const float* Wq     = (const float*)d_in[6];
  const float* Wk     = (const float*)d_in[7];
  const float* WP     = (const float*)d_in[8];
  float* out = (float*)d_out;

  const long NBSD = (long)Bb * Ss * Dd;  // 16,777,216
  float* Qf   = (float*)d_ws;
  float* Kf   = Qf + NBSD;
  float* QK   = Kf + NBSD;
  float* qlog = QK + NBSD;                       // B*H*S = 262,144
  float* klog = qlog + (long)Bb * Hh * Ss;
  float* gq   = klog + (long)Bb * Hh * Ss;
  float* gk   = gq + Bb * Dd;
  __bf16* Abf = (__bf16*)(gk + Bb * Dd);         // 16M bf16 (reused 3x)
  __bf16* Wt  = Abf + NBSD;                      // 1M bf16 (reused 3x)

  const dim3 bb(256);
  const dim3 gg(Dd / 256, (Bb * Ss) / 128);      // (4, 128)
  const dim3 gt(Dd / 32, Dd / 32), bt(32, 8);
  const int M = Bb * Ss;
  const int convG = (int)(NBSD / (256 * 8));     // 8192

  // ---- Q = Q_seq @ WQ ----
  conv_bf16<<<convG, bb, 0, stream>>>(Q_seq, Abf);
  transpose_conv_bf16<<<gt, bt, 0, stream>>>(WQ, Wt, Dd, Dd);
  gemm_bf16_async<false><<<gg, bb, 0, stream>>>(Abf, Wt, nullptr, Qf, M, Dd, Dd);

  // ---- K = K_seq @ WK ----
  conv_bf16<<<convG, bb, 0, stream>>>(K_seq, Abf);
  transpose_conv_bf16<<<gt, bt, 0, stream>>>(WK, Wt, Dd, Dd);
  gemm_bf16_async<false><<<gg, bb, 0, stream>>>(Abf, Wt, nullptr, Kf, M, Dd, Dd);

  // ---- q attention -> global_q ----
  logits_wmma<<<M / 128, bb, 0, stream>>>(Qf, Wq, Q_mask, qlog);
  softmax_rows<<<Bb * Hh, bb, 0, stream>>>(qlog);
  weighted_sum<<<Bb * Hh, bb, 0, stream>>>(qlog, Qf, (long)Ss * Dd, (long)DHd, (long)Dd, gq);

  // ---- QK = Kh * global_q (stored (B,H,S,DH) == the raw-reshape flat view) ----
  qk_scale<<<(int)(NBSD / (4 * 256)), bb, 0, stream>>>(Kf, gq, QK);

  // ---- k attention -> global_k ----
  logits_wmma<<<M / 128, bb, 0, stream>>>(QK, Wk, K_mask, klog);
  softmax_rows<<<Bb * Hh, bb, 0, stream>>>(klog);
  weighted_sum<<<Bb * Hh, bb, 0, stream>>>(klog, QK, (long)Hh * Ss * DHd, (long)Ss * DHd, (long)DHd, gk);

  // ---- out = (Q * gk_bcast) @ WP + Q ----
  scale_conv_bf16<<<convG, bb, 0, stream>>>(Qf, gk, Abf);
  transpose_conv_bf16<<<gt, bt, 0, stream>>>(WP, Wt, Dd, Dd);
  gemm_bf16_async<true><<<gg, bb, 0, stream>>>(Abf, Wt, Qf, out, M, Dd, Dd);
}
